// HierarchicalGNNBlock_50886772523149
// MI455X (gfx1250) — matrix operand order repaired and checked by reading, not compiled, not run on hardware
//
#include <hip/hip_runtime.h>
#include <hip/hip_bf16.h>
#include <stdint.h>

// ---------------------------------------------------------------------------
// CDNA5 (gfx1250) WMMA types
// ---------------------------------------------------------------------------
typedef __attribute__((ext_vector_type(16))) __bf16 v16bf;
typedef __attribute__((ext_vector_type(8)))  float  v8f;

union FragAB { v16bf v; unsigned int u[8]; };
union FragC  { v8f  v; float f[8]; };

// pack two f32 into packed bf16 (truncate): 1 x v_perm_b32
__device__ __forceinline__ unsigned int pack_bf16(float lo, float hi) {
  return __builtin_amdgcn_perm(__float_as_uint(hi), __float_as_uint(lo), 0x07060302u);
}
__device__ __forceinline__ unsigned short bf16_hi(float x) {
  return (unsigned short)(__float_as_uint(x) >> 16);
}
// fast GELU: x * sigmoid(1.702 x)  (~5 VALU: mul, v_exp, add, v_rcp, mul)
__device__ __forceinline__ float gelu_fast(float x) {
  float t = __expf(-1.702f * x);
  return x * __builtin_amdgcn_rcpf(1.0f + t);
}

// B fragment (32x16 bf16, KxN): lane l holds col n=l%16; VGPR v holds k pair
//   v<4: k=2v+8*half ; v>=4: k=16+2(v-4)+8*half  ->  dword v + 8*half of the staged chunk row
// A fragment (16x32 bf16, MxK): lane l holds row m=l%16; VGPR v holds k pair:
//   v<4: k=2v+8*half ; v>=4: k=16+2(v-4)+8*half
//   -> two contiguous 8-float groups at k0 = 8*half and 16+8*half

#define WPAD 17   // dword stride of staged weight chunk row (conflict-free)
#define HPAD 136  // ushort stride of hidden row (68 dwords; 68%64=4 -> conflict-free b128)

// cooperative stage of one 32-deep K chunk of Wt (bf16, transposed [n*K+k])
// into Wlds[n*WPAD + p(j)] (j = k-pair index 0..15, p swaps bits 2 and 3), zero-padded past K.
template <int K>
__device__ __forceinline__ void stage_w(const unsigned short* Wt, int kt,
                                        unsigned int* Wlds, int tid) {
  const int n0 = tid >> 4, j = tid & 15;
  const int k = kt * 32 + 2 * j;           // uniform per thread, loop-invariant
  const unsigned short* gp = Wt + (size_t)n0 * K + k;
  unsigned int* lp = Wlds + n0 * WPAD + ((j & 3) | ((j & 4) << 1) | ((j & 8) >> 1));
#pragma unroll
  for (int r = 0; r < 8; ++r) {
    unsigned int v = 0u;
    if (k < K) v = *(const unsigned int*)gp;
    *lp = v;
    gp += 16 * K;
    lp += 16 * WPAD;
  }
}

// ---------------------------------------------------------------------------
// Fused 2-layer MLP with gather/concat input, bf16 WMMA, GELU, residual.
// ---------------------------------------------------------------------------
struct Mlp2Args {
  const float* src0; const int* idx0;
  const float* src1; const int* idx1;
  const float* src2; const int* idx2;
  const unsigned short* W1t; const float* b1;
  const unsigned short* W2t; const float* b2;
  float* out; const float* res;
  int M; int gelu_out;
};

template <int W0, int W1, int W2>
__global__ __launch_bounds__(256) void mlp2_kernel(Mlp2Args a) {
  constexpr int KIN = W0 + W1 + W2;
  constexpr int NK1 = (KIN + 31) / 32;
  __shared__ unsigned short Hlds[8][16][HPAD];
  __shared__ unsigned int   Wlds[128 * WPAD];

  const int tid  = threadIdx.x;
  const int lane = tid & 31;
  const int wid  = tid >> 5;
  const int half = lane >> 4;
  const int lm   = lane & 15;
  const int tile_base = blockIdx.x * 128 + wid * 16;

  int rowm = tile_base + lm;
  int rc   = rowm < a.M ? rowm : a.M - 1;
  const float* p0 = a.src0 + (size_t)(a.idx0 ? a.idx0[rc] : rc) * W0;
  const float* p1 = (W1 > 0) ? (a.src1 + (size_t)(a.idx1 ? a.idx1[rc] : rc) * W1) : p0;
  const float* p2 = (W2 > 0) ? (a.src2 + (size_t)(a.idx2 ? a.idx2[rc] : rc) * W2) : p0;

  // ---- layer 1: KIN -> 128 ----
  FragC acc[8];
#pragma unroll
  for (int nt = 0; nt < 8; ++nt) {
    float b = a.b1[nt * 16 + lm];
#pragma unroll
    for (int j = 0; j < 8; ++j) acc[nt].f[j] = b;
  }
  for (int kt = 0; kt < NK1; ++kt) {
    __syncthreads();
    stage_w<KIN>(a.W1t, kt, Wlds, tid);

    FragAB A;
#pragma unroll
    for (int g = 0; g < 2; ++g) {
      int k = kt * 32 + g * 16 + 8 * half;
      float4 q0 = make_float4(0.f, 0.f, 0.f, 0.f), q1 = q0;
      if (k < KIN) {  // 8-float group never straddles a 16-aligned segment edge
        const float* pp; int kk;
        if (k < W0)           { pp = p0; kk = k; }
        else if (k < W0 + W1) { pp = p1; kk = k - W0; }
        else                  { pp = p2; kk = k - (W0 + W1); }
        q0 = *(const float4*)(pp + kk);
        q1 = *(const float4*)(pp + kk + 4);
      }
      A.u[4 * g + 0] = pack_bf16(q0.x, q0.y);
      A.u[4 * g + 1] = pack_bf16(q0.z, q0.w);
      A.u[4 * g + 2] = pack_bf16(q1.x, q1.y);
      A.u[4 * g + 3] = pack_bf16(q1.z, q1.w);
    }
    __syncthreads();
#pragma unroll
    for (int nt = 0; nt < 8; ++nt) {
      FragAB B;
      const unsigned int* wrow = &Wlds[(nt * 16 + lm) * WPAD + 8 * half];
#pragma unroll
      for (int v = 0; v < 8; ++v) B.u[v] = wrow[v];
      acc[nt].v = __builtin_amdgcn_wmma_f32_16x16x32_bf16(
          false, A.v, false, B.v, (short)0, acc[nt].v, false, false);
    }
  }
  // gelu -> hidden tile to LDS as bf16
#pragma unroll
  for (int nt = 0; nt < 8; ++nt) {
#pragma unroll
    for (int v = 0; v < 8; ++v) {
      float x = gelu_fast(acc[nt].f[v]);
      Hlds[wid][v + 8 * half][nt * 16 + lm] = bf16_hi(x);
    }
  }

  // ---- layer 2: 128 -> 128 ----
  FragC acc2[8];
#pragma unroll
  for (int nt = 0; nt < 8; ++nt) {
    float b = a.b2[nt * 16 + lm];
#pragma unroll
    for (int j = 0; j < 8; ++j) acc2[nt].f[j] = b;
  }
#pragma unroll
  for (int kt = 0; kt < 4; ++kt) {
    __syncthreads();
    stage_w<128>(a.W2t, kt, Wlds, tid);

    FragAB A;
    {  // two b128 reads from padded hidden row
      const unsigned int* hrow =
          (const unsigned int*)&Hlds[wid][lm][0] + kt * 16 + 4 * half;
#pragma unroll
      for (int v = 0; v < 4; ++v) A.u[v] = hrow[v];
#pragma unroll
      for (int v = 0; v < 4; ++v) A.u[4 + v] = hrow[8 + v];
    }
    __syncthreads();
#pragma unroll
    for (int nt = 0; nt < 8; ++nt) {
      FragAB B;
      const unsigned int* wrow = &Wlds[(nt * 16 + lm) * WPAD + 8 * half];
#pragma unroll
      for (int v = 0; v < 8; ++v) B.u[v] = wrow[v];
      acc2[nt].v = __builtin_amdgcn_wmma_f32_16x16x32_bf16(
          false, A.v, false, B.v, (short)0, acc2[nt].v, false, false);
    }
  }
  // ---- epilogue ----
#pragma unroll
  for (int nt = 0; nt < 8; ++nt) {
#pragma unroll
    for (int v = 0; v < 8; ++v) {
      int r = tile_base + v + 8 * half;
      if (r < a.M) {
        int c = nt * 16 + lm;
        float x = acc2[nt].f[v];
        if (a.gelu_out) x = gelu_fast(x);
        if (a.res) x += a.res[(size_t)r * 128 + c];
        a.out[(size_t)r * 128 + c] = x;
      }
    }
  }
}

// ---------------------------------------------------------------------------
// Fused 3-layer out MLP: concat(nodes[bg0], snodes[bg1]) (256)->128g->128g->1
// ---------------------------------------------------------------------------
struct Out3Args {
  const float* src0; const int* idx0;
  const float* src1; const int* idx1;
  const unsigned short* W1t; const float* b1;
  const unsigned short* W2t; const float* b2;
  const float* W3; const float* b3;
  float* out; int M;
};

__global__ __launch_bounds__(256) void out3_kernel(Out3Args a) {
  constexpr int KIN = 256;
  __shared__ unsigned short Hlds[8][16][HPAD];
  __shared__ unsigned int   Wlds[128 * WPAD];
  const int tid = threadIdx.x, lane = tid & 31, wid = tid >> 5;
  const int half = lane >> 4, lm = lane & 15;
  const int tile_base = blockIdx.x * 128 + wid * 16;

  int rowm = tile_base + lm;
  int rc = rowm < a.M ? rowm : a.M - 1;
  const float* p0 = a.src0 + (size_t)a.idx0[rc] * 128;
  const float* p1 = a.src1 + (size_t)a.idx1[rc] * 128;

  FragC acc[8];
#pragma unroll
  for (int nt = 0; nt < 8; ++nt) {
    float b = a.b1[nt * 16 + lm];
#pragma unroll
    for (int j = 0; j < 8; ++j) acc[nt].f[j] = b;
  }
  for (int kt = 0; kt < 8; ++kt) {
    __syncthreads();
    stage_w<KIN>(a.W1t, kt, Wlds, tid);

    FragAB A;
#pragma unroll
    for (int g = 0; g < 2; ++g) {
      int k = kt * 32 + g * 16 + 8 * half;
      const float* pp = (k < 128) ? p0 : p1;
      int kk = (k < 128) ? k : k - 128;
      float4 q0 = *(const float4*)(pp + kk);
      float4 q1 = *(const float4*)(pp + kk + 4);
      A.u[4 * g + 0] = pack_bf16(q0.x, q0.y);
      A.u[4 * g + 1] = pack_bf16(q0.z, q0.w);
      A.u[4 * g + 2] = pack_bf16(q1.x, q1.y);
      A.u[4 * g + 3] = pack_bf16(q1.z, q1.w);
    }
    __syncthreads();
#pragma unroll
    for (int nt = 0; nt < 8; ++nt) {
      FragAB B;
      const unsigned int* wrow = &Wlds[(nt * 16 + lm) * WPAD + 8 * half];
#pragma unroll
      for (int v = 0; v < 8; ++v) B.u[v] = wrow[v];
      acc[nt].v = __builtin_amdgcn_wmma_f32_16x16x32_bf16(
          false, A.v, false, B.v, (short)0, acc[nt].v, false, false);
    }
  }
#pragma unroll
  for (int nt = 0; nt < 8; ++nt)
#pragma unroll
    for (int v = 0; v < 8; ++v)
      Hlds[wid][v + 8 * half][nt * 16 + lm] = bf16_hi(gelu_fast(acc[nt].f[v]));

  FragC acc2[8];
#pragma unroll
  for (int nt = 0; nt < 8; ++nt) {
    float b = a.b2[nt * 16 + lm];
#pragma unroll
    for (int j = 0; j < 8; ++j) acc2[nt].f[j] = b;
  }
#pragma unroll
  for (int kt = 0; kt < 4; ++kt) {
    __syncthreads();
    stage_w<128>(a.W2t, kt, Wlds, tid);

    FragAB A;
    const unsigned int* hrow =
        (const unsigned int*)&Hlds[wid][lm][0] + kt * 16 + 4 * half;
#pragma unroll
    for (int v = 0; v < 4; ++v) A.u[v] = hrow[v];
#pragma unroll
    for (int v = 0; v < 4; ++v) A.u[4 + v] = hrow[8 + v];
    __syncthreads();
#pragma unroll
    for (int nt = 0; nt < 8; ++nt) {
      FragAB B;
      const unsigned int* wrow = &Wlds[(nt * 16 + lm) * WPAD + 8 * half];
#pragma unroll
      for (int v = 0; v < 8; ++v) B.u[v] = wrow[v];
      acc2[nt].v = __builtin_amdgcn_wmma_f32_16x16x32_bf16(
          false, A.v, false, B.v, (short)0, acc2[nt].v, false, false);
    }
  }
  // layer 3: gelu(H2) . W3 + b3 per row, 16-lane shfl_xor reduction
  float part[8];
#pragma unroll
  for (int v = 0; v < 8; ++v) part[v] = 0.f;
#pragma unroll
  for (int nt = 0; nt < 8; ++nt) {
    float w = a.W3[nt * 16 + lm];
#pragma unroll
    for (int v = 0; v < 8; ++v) part[v] += gelu_fast(acc2[nt].f[v]) * w;
  }
#pragma unroll
  for (int off = 1; off < 16; off <<= 1)
#pragma unroll
    for (int v = 0; v < 8; ++v) part[v] += __shfl_xor(part[v], off, 32);
  if (lm == 0) {
    float bb = a.b3[0];
#pragma unroll
    for (int v = 0; v < 8; ++v) {
      int r = tile_base + v + 8 * half;
      if (r < a.M) a.out[r] = part[v] + bb;
    }
  }
}

// ---------------------------------------------------------------------------
// Helper kernels
// ---------------------------------------------------------------------------
__global__ void wconv_kernel(const float* W, unsigned short* Wt, int K) {
  int idx = blockIdx.x * 256 + threadIdx.x;
  if (idx >= K * 128) return;
  int k = idx >> 7, n = idx & 127;
  Wt[(size_t)n * K + k] = bf16_hi(W[idx]);  // truncate; fp32->bf16
}

__global__ void invl1_kernel(const float* x, float* inv, int M) {
  int i = blockIdx.x * 8 + (threadIdx.x >> 5);
  if (i >= M) return;
  int lane = threadIdx.x & 31;
  float s = 0.f;
  for (int c = lane; c < 128; c += 32) s += fabsf(x[(size_t)i * 128 + c]);
  for (int off = 16; off; off >>= 1) s += __shfl_xor(s, off, 32);
  if (lane == 0) inv[i] = 1.0f / fmaxf(s, 1e-12f);
}

__global__ void scatter_dual_kernel(float* dst, const float* src, const int* ia,
                                    const int* ib, const float* w, int M) {
  int i = blockIdx.x * 8 + (threadIdx.x >> 5);
  if (i >= M) return;
  int lane = threadIdx.x & 31;
  float wt = w ? w[i] : 1.0f;
  int ra = ia[i], rb = ib[i];
  for (int c = lane; c < 128; c += 32) {
    float v = src[(size_t)i * 128 + c] * wt;
    atomicAdd(&dst[(size_t)ra * 128 + c], v);
    atomicAdd(&dst[(size_t)rb * 128 + c], v);
  }
}

__global__ void scatter_gather_kernel(float* dst, const float* src, const int* idxs,
                                      const int* idxd, const float* w,
                                      const float* invl1, int M) {
  int i = blockIdx.x * 8 + (threadIdx.x >> 5);
  if (i >= M) return;
  int lane = threadIdx.x & 31;
  int rs = idxs[i], rd = idxd[i];
  float wt = w[i];
  if (invl1) wt *= invl1[rs];
  for (int c = lane; c < 128; c += 32)
    atomicAdd(&dst[(size_t)rd * 128 + c], src[(size_t)rs * 128 + c] * wt);
}

// ---------------------------------------------------------------------------
// Host launch
// ---------------------------------------------------------------------------
static inline size_t align256(size_t x) { return (x + 255) & ~(size_t)255; }

extern "C" void kernel_launch(void* const* d_in, const int* in_sizes, int n_in,
                              void* d_out, int out_size, void* d_ws, size_t ws_size,
                              hipStream_t stream) {
  (void)n_in; (void)out_size; (void)ws_size;
  const int D = 128, EMB = 16;
  const int N  = in_sizes[0] / D;
  const int E  = in_sizes[1] / D;
  const int S  = in_sizes[2] / EMB;
  const int B  = in_sizes[3];
  const int SE = in_sizes[4];

  const float* nodes_in = (const float*)d_in[0];
  const float* edges_in = (const float*)d_in[1];
  const float* semb     = (const float*)d_in[2];
  const float* bw       = (const float*)d_in[3];
  const float* sw       = (const float*)d_in[4];
  const int* g0  = (const int*)d_in[5]; const int* g1  = g0 + E;
  const int* bg0 = (const int*)d_in[6]; const int* bg1 = bg0 + B;
  const int* sg0 = (const int*)d_in[7]; const int* sg1 = sg0 + SE;

  char* ws = (char*)d_ws;
  size_t off = 0;
  auto carve = [&](size_t bytes) { char* p = ws + off; off = align256(off + bytes); return p; };
  float* nodes  = (float*)carve((size_t)N * D * 4);
  float* edges  = (float*)carve((size_t)E * D * 4);
  float* snodes = (float*)carve((size_t)S * D * 4);
  float* sedges = (float*)carve((size_t)SE * D * 4);
  float* aggN   = (float*)carve((size_t)N * D * 4);
  float* aggS   = (float*)carve((size_t)S * D * 4);
  float* invl1  = (float*)carve((size_t)N * 4);

  // params flatten order (insertion order): d_in[8..]
  struct WDesc { int idx; int K; };
  const WDesc wl[22] = {
    {8, 144}, {10, 128},                        // snode_enc W0,W1
    {12, 256}, {14, 128},                       // sedge_enc
    {16, 256}, {18, 128},                       // out L0,L1
    {22, 256}, {24, 128}, {26, 384}, {28, 128}, // cell0 node, edge
    {30, 256}, {32, 128}, {34, 384}, {36, 128}, // cell0 snode, sedge
    {38, 256}, {40, 128}, {42, 384}, {44, 128}, // cell1 node, edge
    {46, 256}, {48, 128}, {50, 384}, {52, 128}, // cell1 snode, sedge
  };
  unsigned short* wt[22];
  for (int i = 0; i < 22; ++i) wt[i] = (unsigned short*)carve((size_t)wl[i].K * 128 * 2);
  for (int i = 0; i < 22; ++i) {
    int total = wl[i].K * 128;
    wconv_kernel<<<(total + 255) / 256, 256, 0, stream>>>(
        (const float*)d_in[wl[i].idx], wt[i], wl[i].K);
  }

  hipMemcpyAsync(nodes, nodes_in, (size_t)N * D * 4, hipMemcpyDeviceToDevice, stream);
  hipMemcpyAsync(edges, edges_in, (size_t)E * D * 4, hipMemcpyDeviceToDevice, stream);

  auto rows_grid = [](int rows) { return (rows + 127) / 128; };
  auto item_grid = [](int m) { return (m + 7) / 8; };

  // ---- supernode init ----
  invl1_kernel<<<item_grid(N), 256, 0, stream>>>(nodes, invl1, N);
  hipMemsetAsync(snodes, 0, (size_t)S * D * 4, stream);
  scatter_gather_kernel<<<item_grid(B), 256, 0, stream>>>(snodes, nodes, bg0, bg1, bw, invl1, B);
  {
    Mlp2Args a{semb, nullptr, snodes, nullptr, nullptr, nullptr,
               wt[0], (const float*)d_in[9], wt[1], (const float*)d_in[11],
               snodes, nullptr, S, 1};
    mlp2_kernel<16, 128, 0><<<rows_grid(S), 256, 0, stream>>>(a);
  }
  {
    Mlp2Args a{snodes, sg0, snodes, sg1, nullptr, nullptr,
               wt[2], (const float*)d_in[13], wt[3], (const float*)d_in[15],
               sedges, nullptr, SE, 1};
    mlp2_kernel<128, 128, 0><<<rows_grid(SE), 256, 0, stream>>>(a);
  }

  // ---- message passing cells ----
  for (int c = 0; c < 2; ++c) {
    const int wb = 6 + c * 8;
    const int pb = 22 + c * 16;
    {  // edge update (E x [384->128->128], residual)
      Mlp2Args a{nodes, g0, nodes, g1, edges, nullptr,
                 wt[wb + 2], (const float*)d_in[pb + 5], wt[wb + 3], (const float*)d_in[pb + 7],
                 edges, edges, E, 0};
      mlp2_kernel<128, 128, 128><<<rows_grid(E), 256, 0, stream>>>(a);
    }
    {  // sedge update
      Mlp2Args a{snodes, sg0, snodes, sg1, sedges, nullptr,
                 wt[wb + 6], (const float*)d_in[pb + 13], wt[wb + 7], (const float*)d_in[pb + 15],
                 sedges, sedges, SE, 0};
      mlp2_kernel<128, 128, 128><<<rows_grid(SE), 256, 0, stream>>>(a);
    }
    hipMemsetAsync(aggN, 0, (size_t)N * D * 4, stream);
    scatter_dual_kernel<<<item_grid(E), 256, 0, stream>>>(aggN, edges, g1, g0, nullptr, E);
    scatter_gather_kernel<<<item_grid(B), 256, 0, stream>>>(aggN, snodes, bg1, bg0, bw, nullptr, B);
    hipMemsetAsync(aggS, 0, (size_t)S * D * 4, stream);
    scatter_dual_kernel<<<item_grid(SE), 256, 0, stream>>>(aggS, sedges, sg1, sg0, sw, SE);
    invl1_kernel<<<item_grid(N), 256, 0, stream>>>(nodes, invl1, N);
    scatter_gather_kernel<<<item_grid(B), 256, 0, stream>>>(aggS, nodes, bg0, bg1, bw, invl1, B);
    {  // node update
      Mlp2Args a{nodes, nullptr, aggN, nullptr, nullptr, nullptr,
                 wt[wb + 0], (const float*)d_in[pb + 1], wt[wb + 1], (const float*)d_in[pb + 3],
                 nodes, nodes, N, 0};
      mlp2_kernel<128, 128, 0><<<rows_grid(N), 256, 0, stream>>>(a);
    }
    {  // snode update
      Mlp2Args a{snodes, nullptr, aggS, nullptr, nullptr, nullptr,
                 wt[wb + 4], (const float*)d_in[pb + 9], wt[wb + 5], (const float*)d_in[pb + 11],
                 snodes, snodes, S, 0};
      mlp2_kernel<128, 128, 0><<<rows_grid(S), 256, 0, stream>>>(a);
    }
  }

  // ---- output head ----
  {
    Out3Args a{nodes, bg0, snodes, bg1,
               wt[4], (const float*)d_in[17], wt[5], (const float*)d_in[19],
               (const float*)d_in[20], (const float*)d_in[21],
               (float*)d_out, B};
    out3_kernel<<<rows_grid(B), 256, 0, stream>>>(a);
  }
}
